// InnerConvexViolationProjection_35648228557556
// MI455X (gfx1250) — matrix operand order repaired and checked by reading, not compiled, not run on hardware
//
#include <hip/hip_runtime.h>

#define N 512
#define NPROB 256      // B*S = 32*8
#define K_ITERS 30
#define P_ITERS 5
#define RHO 1e-12f

typedef __attribute__((ext_vector_type(16))) __bf16 v16bf;
typedef __attribute__((ext_vector_type(8)))  __bf16 v8bf;
typedef __attribute__((ext_vector_type(8)))  float  v8f;

// ---------------------------------------------------------------------------
// Kernel 1: A (fp32, row-major) -> Ab (bf16, row-major) and Atb (bf16, A^T
// row-major). One 32x32 tile per block, transpose via LDS.
// ---------------------------------------------------------------------------
__global__ __launch_bounds__(256) void convert_transpose_kernel(
    const float* __restrict__ A, __bf16* __restrict__ Ab, __bf16* __restrict__ Atb) {
  __shared__ float tile[32][33];
  const int p  = blockIdx.z;
  const int r0 = blockIdx.y * 32;
  const int c0 = blockIdx.x * 32;
  const size_t base = (size_t)p * N * N;
  const int tx = threadIdx.x, ty = threadIdx.y;
#pragma unroll
  for (int i = ty; i < 32; i += 8) {
    float v = A[base + (size_t)(r0 + i) * N + (c0 + tx)];
    Ab[base + (size_t)(r0 + i) * N + (c0 + tx)] = (__bf16)v;
    tile[i][tx] = v;
  }
  __syncthreads();
#pragma unroll
  for (int i = ty; i < 32; i += 8) {
    Atb[base + (size_t)(c0 + i) * N + (r0 + tx)] = (__bf16)tile[tx][i];
  }
}

// ---------------------------------------------------------------------------
// 16 rows of a row-major bf16 matrix times a 512-vector held in LDS.
//
// A-fragment, contiguous k order (B matches): lane l<16 holds row M=l, K
// [k0..k0+16); lane l+16 holds the same row, K [k0+16..k0+32).
//
// B-fragment: the vector chunk broadcast into ALL 16 columns (per-lane
// address depends only on lane>>4 -> conflict-free LDS broadcast, no
// predication/zero-fill). Every column of D is an identical copy of the
// matvec result; lanes 0/16 extract it.
//
// Modulo-2 software pipeline: two buffer sets in alternating phases. Each
// buffer's next definition comes AFTER its last use, so the allocator reuses
// the same physical registers (no rotation copies, no WMMA->VALU WAR NOPs),
// and each WMMA's operands were loaded one phase earlier (distance-1
// partial-count waits instead of full drains).
// ---------------------------------------------------------------------------
__device__ __forceinline__ v8f mv_rows16(const __bf16* __restrict__ Mrow,
                                         const __bf16* vlds, unsigned lane) {
  v8f c = {};
  const __bf16* rp   = Mrow + (size_t)(lane & 15u) * N + ((lane >> 4) << 4);
  const __bf16* vsrc = vlds + ((lane >> 4) << 4);   // +0 or +16

  union AF { v16bf v; v8bf h[2]; };
  AF a0;  v16bf b0;
  a0.h[0] = *reinterpret_cast<const v8bf*>(rp);          // global b128
  a0.h[1] = *reinterpret_cast<const v8bf*>(rp + 8);     // global b128
  b0      = *reinterpret_cast<const v16bf*>(vsrc);       // 2x ds b128

#pragma unroll 1
  for (int kk = 0; kk < 14; kk += 2) {
    AF a1;  v16bf b1;
    a1.h[0] = *reinterpret_cast<const v8bf*>(rp + (kk + 1) * 32);
    a1.h[1] = *reinterpret_cast<const v8bf*>(rp + (kk + 1) * 32 + 8);
    b1      = *reinterpret_cast<const v16bf*>(vsrc + (kk + 1) * 32);
    c = __builtin_amdgcn_wmma_f32_16x16x32_bf16(
        false, a0.v, false, b0, (short)0, c, false, false);
    a0.h[0] = *reinterpret_cast<const v8bf*>(rp + (kk + 2) * 32);
    a0.h[1] = *reinterpret_cast<const v8bf*>(rp + (kk + 2) * 32 + 8);
    b0      = *reinterpret_cast<const v16bf*>(vsrc + (kk + 2) * 32);
    c = __builtin_amdgcn_wmma_f32_16x16x32_bf16(
        false, a1.v, false, b1, (short)0, c, false, false);
  }
  // epilogue pair (kk == 14): no further prefetch
  {
    AF a1;  v16bf b1;
    a1.h[0] = *reinterpret_cast<const v8bf*>(rp + 15 * 32);
    a1.h[1] = *reinterpret_cast<const v8bf*>(rp + 15 * 32 + 8);
    b1      = *reinterpret_cast<const v16bf*>(vsrc + 15 * 32);
    c = __builtin_amdgcn_wmma_f32_16x16x32_bf16(
        false, a0.v, false, b0, (short)0, c, false, false);
    c = __builtin_amdgcn_wmma_f32_16x16x32_bf16(
        false, a1.v, false, b1, (short)0, c, false, false);
  }
  return c;
}

__device__ __forceinline__ float blockSum(float v, float* red, int tid) {
#pragma unroll
  for (int off = 16; off > 0; off >>= 1) v += __shfl_down(v, off, 32);
  __syncthreads();                       // protect `red` from previous use
  if ((tid & 31) == 0) red[tid >> 5] = v;
  __syncthreads();
  float s = 0.f;
#pragma unroll
  for (int i = 0; i < 8; ++i) s += red[i];
  return s;
}

// Forward half-pass: u = A*vec (optionally relu(A*vec - b)); 64 rows per wave.
__device__ __forceinline__ void forward_pass(const __bf16* __restrict__ Ap,
                                             const float* bl, __bf16* ub,
                                             const __bf16* vecb, int wave,
                                             unsigned lane, bool proj) {
#pragma unroll 1
  for (int t = 0; t < 4; ++t) {
    const int rowBase = wave * 64 + t * 16;
    const __bf16* Mrow = Ap + (size_t)rowBase * N;
    if (t < 3)   // warm next tile's rows while this tile computes
      __builtin_prefetch(Mrow + 16 * N + (size_t)(lane & 15u) * N, 0, 3);
    v8f c = mv_rows16(Mrow, vecb, lane);
    if (lane == 0u || lane == 16u) {
      const int r0 = rowBase + (int)((lane >> 4) << 3);
      union { v8bf v; __bf16 e[8]; } s;
#pragma unroll
      for (int j = 0; j < 8; ++j) {
        float r = c[j];
        if (proj) { r -= bl[r0 + j]; r = fmaxf(r, 0.f); }
        s.e[j] = (__bf16)r;
      }
      *reinterpret_cast<v8bf*>(ub + r0) = s.v;   // one ds_store_b128
    }
  }
}

// Transpose half-pass (power): w = A^T * u, written as f32.
__device__ __forceinline__ void transpose_pass_power(const __bf16* __restrict__ Atp,
                                                     float* wf, const __bf16* ub,
                                                     int wave, unsigned lane) {
#pragma unroll 1
  for (int t = 0; t < 4; ++t) {
    const int colBase = wave * 64 + t * 16;
    const __bf16* Mrow = Atp + (size_t)colBase * N;
    if (t < 3)
      __builtin_prefetch(Mrow + 16 * N + (size_t)(lane & 15u) * N, 0, 3);
    v8f c = mv_rows16(Mrow, ub, lane);
    if (lane == 0u || lane == 16u) {
      const int c0 = colBase + (int)((lane >> 4) << 3);
      *reinterpret_cast<v8f*>(wf + c0) = c;      // two ds_store_b128
    }
  }
}

// Transpose half-pass (projection): x = (x - eta * A^T relu(Ax-b)) * mask.
__device__ __forceinline__ void transpose_pass_proj(const __bf16* __restrict__ Atp,
                                                    float* xf, const float* ml,
                                                    float eta, const __bf16* ub,
                                                    int wave, unsigned lane) {
#pragma unroll 1
  for (int t = 0; t < 4; ++t) {
    const int colBase = wave * 64 + t * 16;
    const __bf16* Mrow = Atp + (size_t)colBase * N;
    if (t < 3)
      __builtin_prefetch(Mrow + 16 * N + (size_t)(lane & 15u) * N, 0, 3);
    v8f c = mv_rows16(Mrow, ub, lane);
    if (lane == 0u || lane == 16u) {
      const int c0 = colBase + (int)((lane >> 4) << 3);
      v8f x = *reinterpret_cast<const v8f*>(xf + c0);
      v8f m = *reinterpret_cast<const v8f*>(ml + c0);
      v8f xn = (x - eta * c) * m;
      *reinterpret_cast<v8f*>(xf + c0) = xn;
    }
  }
}

// ---------------------------------------------------------------------------
// Kernel 2: one workgroup (8 wave32) per problem. Power iteration for eta,
// then 30 fused projection iterations, all state in LDS; A streamed from L2.
// ---------------------------------------------------------------------------
__global__ __launch_bounds__(256) void solve_kernel(
    const float* __restrict__ x_in, const float* __restrict__ b_in,
    const float* __restrict__ mask_in, const float* __restrict__ v0_in,
    const __bf16* __restrict__ Ab, const __bf16* __restrict__ Atb,
    float* __restrict__ x_out) {
  __shared__ __align__(32) float  xf[N], bl[N], ml[N], vf[N], wf[N];
  __shared__ __align__(32) __bf16 vecb[N], ub[N];
  __shared__ float red[8];

  const int tid  = threadIdx.x;
  const unsigned lane = (unsigned)(tid & 31);
  const int wave = tid >> 5;
  const size_t pb = (size_t)blockIdx.x * N;
  const __bf16* Ap  = Ab  + pb * N;
  const __bf16* Atp = Atb + pb * N;

  // Load per-problem state; start power-iteration vector from v0.
  float acc = 0.f;
  for (int i = tid; i < N; i += 256) {
    xf[i] = x_in[pb + i];
    bl[i] = b_in[pb + i];
    ml[i] = mask_in[pb + i];
    float v0 = v0_in[pb + i];
    vf[i] = v0;
    acc += v0 * v0;
  }
  float nrm = sqrtf(blockSum(acc, red, tid));
  for (int i = tid; i < N; i += 256) {
    float t = vf[i] / (nrm + RHO);
    vf[i] = t;
    vecb[i] = (__bf16)t;
  }
  __syncthreads();

  // 5 power iterations on A^T A (each a fused Av / A^T(Av) pass).
  for (int it = 0; it < P_ITERS; ++it) {
    forward_pass(Ap, bl, ub, vecb, wave, lane, false);
    __syncthreads();
    transpose_pass_power(Atp, wf, ub, wave, lane);
    __syncthreads();
    float a2 = 0.f;
    for (int i = tid; i < N; i += 256) { float w = wf[i]; a2 += w * w; }
    float nw = sqrtf(blockSum(a2, red, tid));
    for (int i = tid; i < N; i += 256) {
      float t = wf[i] / (nw + RHO);
      vf[i] = t;
      vecb[i] = (__bf16)t;
    }
    __syncthreads();
  }

  // eta = 1 / (||A v||^2 + rho); note ||Av||^2 = v . (A^T A v).
  forward_pass(Ap, bl, ub, vecb, wave, lane, false);
  __syncthreads();
  transpose_pass_power(Atp, wf, ub, wave, lane);
  __syncthreads();
  float d = 0.f;
  for (int i = tid; i < N; i += 256) d += vf[i] * wf[i];
  const float eta = 1.f / (blockSum(d, red, tid) + RHO);

  // 30 projection iterations, one fused pass over A + A^T each.
  for (int it = 0; it < K_ITERS; ++it) {
    for (int i = tid; i < N; i += 256) vecb[i] = (__bf16)xf[i];
    __syncthreads();
    forward_pass(Ap, bl, ub, vecb, wave, lane, true);   // ub = relu(Ax-b)
    __syncthreads();
    transpose_pass_proj(Atp, xf, ml, eta, ub, wave, lane);
    __syncthreads();
  }

  for (int i = tid; i < N; i += 256) x_out[pb + i] = xf[i];
}

// ---------------------------------------------------------------------------
extern "C" void kernel_launch(void* const* d_in, const int* in_sizes, int n_in,
                              void* d_out, int out_size, void* d_ws, size_t ws_size,
                              hipStream_t stream) {
  const float* x    = (const float*)d_in[0];
  const float* A    = (const float*)d_in[1];
  const float* b    = (const float*)d_in[2];
  const float* mask = (const float*)d_in[3];
  const float* v0   = (const float*)d_in[4];
  float* out = (float*)d_out;

  __bf16* Ab  = (__bf16*)d_ws;
  __bf16* Atb = Ab + (size_t)NPROB * N * N;   // 134 MB each in ws

  convert_transpose_kernel<<<dim3(N / 32, N / 32, NPROB), dim3(32, 8), 0, stream>>>(
      A, Ab, Atb);
  solve_kernel<<<NPROB, 256, 0, stream>>>(x, b, mask, v0, Ab, Atb, out);
}
